// SNNBaseline_678604833394
// MI455X (gfx1250) — matrix operand order repaired and checked by reading, not compiled, not run on hardware
//
#include <hip/hip_runtime.h>
#include <hip/hip_bf16.h>

typedef _Float16 v16h __attribute__((ext_vector_type(16)));
typedef _Float16 h8   __attribute__((ext_vector_type(8)));
typedef float    v8f  __attribute__((ext_vector_type(8)));

#define T_STEPS 1000
#define BATCH   256
#define FEAT    64
#define H1      256
#define H2      256
#define BETA1   0.81873075307798185867f   /* exp(-1/5)  */
#define BETA2   0.90483741803595957316f   /* exp(-1/10) */

// ---- dynamic LDS partition (bytes) ----
#define W2H_OFF   0                        // W2^T B-fragments f16: 8*16*32*16 halves = 128 KB
#define XB_OFF    (W2H_OFF + 128*1024)     // x_t row-major f16:  16*64  = 2 KB
#define SPK_OFF   (XB_OFF + 2*1024)        // spk1 row-major f16: 16*256 = 8 KB
#define OUTB_OFF  (SPK_OFF + 8*1024)       // mem2_avg f32:       16*256 = 16 KB
#define RED_OFF   (OUTB_OFF + 16*1024)     // 2 reduce slots
#define SMEM_BYTES (RED_OFF + 64)

static __device__ inline v8f wmma_f16(v16h a, v16h b, v8f c) {
    return __builtin_amdgcn_wmma_f32_16x16x32_f16(
        /*neg_a=*/false, a, /*neg_b=*/false, b,
        /*c_mod=*/(short)0, c, /*reuse_a=*/false, /*reuse_b=*/false);
}

static __device__ inline v8f splat8(float v) {
    v8f r;
#pragma unroll
    for (int i = 0; i < 8; ++i) r[i] = v;
    return r;
}

// Build a 16-bit fragment from two 16-byte-aligned LDS chunks (2x ds_load_b128).
static __device__ inline v16h make_frag(const _Float16* p_lo, const _Float16* p_hi) {
    h8 lo = *(const h8*)p_lo;
    h8 hi = *(const h8*)p_hi;
    v16h f;
#pragma unroll
    for (int i = 0; i < 8; ++i) { f[i] = lo[i]; f[i + 8] = hi[i]; }
    return f;
}

// Load 16 contiguous f32 from global, convert to a register-resident f16 fragment.
static __device__ inline v16h load_wfrag(const float* __restrict__ src) {
    v16h f;
#pragma unroll
    for (int i = 0; i < 16; i += 4) {
        float4 v = *(const float4*)(src + i);
        f[i + 0] = (_Float16)v.x; f[i + 1] = (_Float16)v.y;
        f[i + 2] = (_Float16)v.z; f[i + 3] = (_Float16)v.w;
    }
    return f;
}

// Convert 16 contiguous f32 -> 16 contiguous f16 in LDS.
static __device__ inline void cvt_store16(_Float16* dst, const float* src) {
#pragma unroll
    for (int i = 0; i < 16; i += 4) {
        float4 v = *(const float4*)(src + i);
        dst[i + 0] = (_Float16)v.x; dst[i + 1] = (_Float16)v.y;
        dst[i + 2] = (_Float16)v.z; dst[i + 3] = (_Float16)v.w;
    }
}

// LIF update on one C-fragment: mem = beta*mem + cur - s_prev; s = (mem > 1).
static __device__ inline void lif_update(v8f& mem, v8f& s, const v8f& cur, float beta) {
#pragma unroll
    for (int r = 0; r < 8; ++r) {
        float m = beta * mem[r] + cur[r] - s[r];
        mem[r] = m;
        s[r] = (m > 1.0f) ? 1.0f : 0.0f;
    }
}

__global__ __launch_bounds__(256, 1)
void snn_main(const float* __restrict__ spikes,
              const float* __restrict__ W1, const float* __restrict__ b1,
              const float* __restrict__ W2, const float* __restrict__ b2,
              const float* __restrict__ Wr, const float* __restrict__ br,
              float* __restrict__ out, float* __restrict__ ws) {
    extern __shared__ char smem[];
    _Float16* w2h  = (_Float16*)(smem + W2H_OFF);
    _Float16* xb   = (_Float16*)(smem + XB_OFF);
    _Float16* spkb = (_Float16*)(smem + SPK_OFF);
    float*    outb = (float*)(smem + OUTB_OFF);
    float*    red  = (float*)(smem + RED_OFF);

    const int tid  = threadIdx.x;
    const int lane = tid & 31;
    const int wv   = tid >> 5;          // 8 waves
    const int lm   = lane & 15;
    const int hi16 = lane >> 4;         // 0 or 1
    const int blk  = blockIdx.x;        // 16 batch tiles of M=16

    // ---- pre-swizzle W2^T into LDS in B-fragment order (one-time) ----
    // chunk c -> (kt, nt, lane): per-lane 16 halves, k = kt*32 + (lane<16?0:16) + j
    for (int c = tid; c < 8 * 16 * 32; c += 256) {
        int l  = c & 31, fr = c >> 5;
        int nt = fr & 15, kt = fr >> 4;
        int n  = nt * 16 + (l & 15);
        int kb = kt * 32 + ((l < 16) ? 0 : 16);
        cvt_store16(w2h + c * 16, W2 + n * H1 + kb);
    }

    // per-wave column tiles and biases
    const int n0 = wv * 2, n1 = wv * 2 + 1;
    const float bias1_0 = b1[n0 * 16 + lm], bias1_1 = b1[n1 * 16 + lm];
    const float bias2_0 = b2[n0 * 16 + lm], bias2_1 = b2[n1 * 16 + lm];

    // ---- register-resident W1^T B-fragments (4 x v16h = 32 VGPRs) ----
    v16h w1f[2][2];
#pragma unroll
    for (int kt = 0; kt < 2; ++kt) {
        const int kb = kt * 32 + hi16 * 16;
        w1f[kt][0] = load_wfrag(W1 + (n0 * 16 + lm) * FEAT + kb);
        w1f[kt][1] = load_wfrag(W1 + (n1 * 16 + lm) * FEAT + kb);
    }

    // persistent register state (C-fragment layout)
    v8f mem1a = splat8(0.f), mem1b = splat8(0.f);
    v8f mem2a = splat8(0.f), mem2b = splat8(0.f);
    v8f msa   = splat8(0.f), msb   = splat8(0.f);
    v8f s1a   = splat8(0.f), s1b   = splat8(0.f);
    v8f s2a   = splat8(0.f), s2b   = splat8(0.f);
    v8f cnt1v = splat8(0.f), cnt2v = splat8(0.f);

    const int kba = hi16 * 16;                          // A-fragment K base
    const int srow = tid >> 4, scol = (tid & 15) * 4;   // x staging: 4 floats/thread

    for (int t = 0; t < T_STEPS; ++t) {
        // ---- stage x_t tile [16 x 64] into LDS as f16 ----
        {
            const float* gp = spikes + (size_t)t * (BATCH * FEAT)
                                     + (size_t)(blk * 16 + srow) * FEAT + scol;
            float4 v = *(const float4*)gp;
            _Float16* dp = xb + srow * FEAT + scol;
            dp[0] = (_Float16)v.x; dp[1] = (_Float16)v.y;
            dp[2] = (_Float16)v.z; dp[3] = (_Float16)v.w;
            if (t + 1 < T_STEPS)
                __builtin_prefetch(gp + BATCH * FEAT, 0, 0);   // global_prefetch_b8
        }
        __syncthreads();

        // ---- layer 1: cur1 = x @ W1^T + b1 (K = 64 -> 2 WMMA k-steps) ----
        v8f c0 = splat8(bias1_0), c1 = splat8(bias1_1);
#pragma unroll
        for (int kt = 0; kt < 2; ++kt) {
            const _Float16* ap = xb + lm * FEAT + kt * 32 + kba;
            v16h a = make_frag(ap, ap + 8);
            c0 = wmma_f16(a, w1f[kt][0], c0);
            c1 = wmma_f16(a, w1f[kt][1], c1);
        }
        lif_update(mem1a, s1a, c0, BETA1);
        lif_update(mem1b, s1b, c1, BETA1);
        cnt1v += s1a; cnt1v += s1b;

        // publish spk1 row-major [16 x 256] f16
        {
            const int col0 = n0 * 16 + lm, col1 = n1 * 16 + lm;
#pragma unroll
            for (int r = 0; r < 8; ++r) {
                int row = r + hi16 * 8;
                spkb[row * H1 + col0] = (_Float16)s1a[r];
                spkb[row * H1 + col1] = (_Float16)s1b[r];
            }
        }
        __syncthreads();

        // ---- layer 2: cur2 = spk1 @ W2^T + b2 (K = 256 -> 8 WMMA k-steps) ----
        c0 = splat8(bias2_0); c1 = splat8(bias2_1);
#pragma unroll
        for (int kt = 0; kt < 8; ++kt) {
            const _Float16* ap = spkb + lm * H1 + kt * 32 + kba;
            v16h a = make_frag(ap, ap + 8);
            v16h bf0 = *(const v16h*)(w2h + ((kt * 16 + n0) * 32 + lane) * 16);
            v16h bf1 = *(const v16h*)(w2h + ((kt * 16 + n1) * 32 + lane) * 16);
            c0 = wmma_f16(a, bf0, c0);
            c1 = wmma_f16(a, bf1, c1);
        }
        lif_update(mem2a, s2a, c0, BETA2);
        lif_update(mem2b, s2b, c1, BETA2);
        cnt2v += s2a; cnt2v += s2b;
        msa += mem2a;
        msb += mem2b;
    }

    // ---- readout: out = (mem2_sum / T) @ Wr^T + br ----
    {
        const float invT = 1.0f / (float)T_STEPS;
        const int col0 = n0 * 16 + lm, col1 = n1 * 16 + lm;
#pragma unroll
        for (int r = 0; r < 8; ++r) {
            int row = r + hi16 * 8;
            outb[row * H2 + col0] = msa[r] * invT;
            outb[row * H2 + col1] = msb[r] * invT;
        }
    }
    if (tid == 0) { red[0] = 0.f; red[1] = 0.f; }
    __syncthreads();

    if (tid < 32) {
        int m = tid & 15, o = tid >> 4;
        float acc = br[o];
        const float* wr = Wr + o * H2;
        for (int h = 0; h < H2; ++h) acc += outb[m * H2 + h] * wr[h];
        out[(blk * 16 + m) * 2 + o] = acc;
    }

    // ---- per-block spike-count partials ----
    float cnt1 = 0.f, cnt2 = 0.f;
#pragma unroll
    for (int r = 0; r < 8; ++r) { cnt1 += cnt1v[r]; cnt2 += cnt2v[r]; }
    atomicAdd(&red[0], cnt1);
    atomicAdd(&red[1], cnt2);
    __syncthreads();
    if (tid == 0) {
        ws[blk * 2 + 0] = red[0];
        ws[blk * 2 + 1] = red[1];
    }
}

__global__ void snn_finalize(const float* __restrict__ ws, float* __restrict__ out) {
    if (threadIdx.x == 0) {
        float s1 = 0.f, s2 = 0.f;
        for (int i = 0; i < 16; ++i) { s1 += ws[2 * i]; s2 += ws[2 * i + 1]; }
        const float sc1 = 1.0f / ((float)T_STEPS * (float)BATCH * (float)H1);
        const float sc2 = 1.0f / ((float)T_STEPS * (float)BATCH * (float)H2);
        out[512] = s1 * sc1;
        out[513] = s2 * sc2;
    }
}

extern "C" void kernel_launch(void* const* d_in, const int* in_sizes, int n_in,
                              void* d_out, int out_size, void* d_ws, size_t ws_size,
                              hipStream_t stream) {
    const float* spikes = (const float*)d_in[0];
    const float* W1 = (const float*)d_in[1];
    const float* b1 = (const float*)d_in[2];
    const float* W2 = (const float*)d_in[3];
    const float* b2 = (const float*)d_in[4];
    const float* Wr = (const float*)d_in[5];
    const float* br = (const float*)d_in[6];
    float* out = (float*)d_out;
    float* ws  = (float*)d_ws;

    snn_main<<<dim3(BATCH / 16), dim3(256), SMEM_BYTES, stream>>>(
        spikes, W1, b1, W2, b2, Wr, br, out, ws);
    snn_finalize<<<dim3(1), dim3(32), 0, stream>>>(ws, out);
}
